// MeshPointDiscriminator_12790412607774
// MI455X (gfx1250) — matrix operand order repaired and checked by reading, not compiled, not run on hardware
//
#include <hip/hip_runtime.h>
#include <math.h>

typedef __attribute__((ext_vector_type(16))) _Float16 v16h;
typedef __attribute__((ext_vector_type(8)))  _Float16 v8h;
typedef __attribute__((ext_vector_type(4)))  _Float16 v4h;
typedef __attribute__((ext_vector_type(8)))  float    v8f;

#define NT 128         // N (position) tile per block
#define BT_STRIDE 40   // halves per LDS row: 32 + 8 pad (80B stride -> bank spread)

// ---------------------------------------------------------------------------
// Weight convert: f32 (rows x Kvalid) -> f16 (rows x Kpad), zero padded K tail
// ---------------------------------------------------------------------------
__global__ void cvt_kernel(const float* __restrict__ w, _Float16* __restrict__ o,
                           int Kv, int Kp, int rows) {
    int i = blockIdx.x * blockDim.x + threadIdx.x;
    if (i >= rows * Kp) return;
    int r = i / Kp, k = i - r * Kp;
    o[i] = (k < Kv) ? (_Float16)w[(size_t)r * Kv + k] : (_Float16)0.0f;
}

// ---------------------------------------------------------------------------
// mesh_conv GEMM: out[b,o,f] = sum_{c,j} w[o,c*4+j] * G(x)[c,f,j]
//   PREOP   : input value v -> sc[c]*max(v,0)+sh[c]  (relu + BN affine)
//   PARTIAL : Cin*4 < Kpad. Channel index is CLAMPED (no zero-fill): the K tail
//             of w16 is zero-padded, so clamped-channel garbage contributes 0.
//   RESID   : out = relu(acc + res)                  (residual epilogue)
// grid: (ceil(F/128), Cout/64, 8batch), block 256 (8 wave32)
// Each wave owns a 32(M) x 32(N) output tile = 4 f32 WMMA accumulators.
// The K loop is peeled: in-loop gathers are unconditional so the WMMA wait
// threshold stays partial (gathers in flight through the matrix ops).
// ---------------------------------------------------------------------------
template <int PREOP, int PARTIAL, int RESID>
__global__ __launch_bounds__(256) void conv_kernel(
    const float* __restrict__ x, const int* __restrict__ nbr,
    const _Float16* __restrict__ w16, float* __restrict__ out,
    const float* __restrict__ res, const float* __restrict__ sc,
    const float* __restrict__ sh,
    int Cin, int Cout, int F, int Kpad)
{
    __shared__ _Float16 Bt[2][NT * BT_STRIDE];  // 2 x 128 x 40 halves = 20 KB

    const int tid    = threadIdx.x;
    const int b      = blockIdx.z;
    const int n0     = blockIdx.x * NT;
    const int lane   = tid & 31;
    const int l16    = lane & 15;
    const int half16 = lane >> 4;
    // wave-uniform scalars (provably uniform for the backend)
    const int waveu  = __builtin_amdgcn_readfirstlane(tid >> 5);
    const int cb     = __builtin_amdgcn_readfirstlane(tid >> 7);   // 0 or 1
    const int m_base = blockIdx.y * 64 + (waveu >> 2) * 32;  // 2 M-tiles per wave
    const int n_sub  = (waveu & 3) * 32;                     // 2 N-frags per wave

    // ---- fixed per-thread gather geometry: one position, 4 channels/chunk ----
    const int n_loc = tid & (NT - 1);
    int f = n0 + n_loc; if (f >= F) f = F - 1;
    const int i1 = nbr[f * 3 + 0];
    const int i2 = nbr[f * 3 + 1];
    const int i3 = nbr[f * 3 + 2];
    const float* xbat = x + (size_t)b * Cin * F;             // scalar batch base

    float rx[4], r1[4], r2[4], r3[4];

    // issue gather loads for one K-chunk (channels chunk*8 + cb + {0,2,4,6});
    // straight-line: PARTIAL only clamps the (scalar) channel index
    auto gather = [&](int chunk) {
#pragma unroll
        for (int it = 0; it < 4; ++it) {
            int c = (chunk << 3) + cb + it * 2;              // scalar
            if (PARTIAL) c = (c < Cin) ? c : (Cin - 1);      // scalar clamp
            const float* xb = xbat + (size_t)c * F;          // scalar row
            rx[it] = xb[f];  r1[it] = xb[i1];
            r2[it] = xb[i2]; r3[it] = xb[i3];
        }
    };
    // G-math + f16 LDS commit (consumes gathered registers; BN coeffs via SMEM)
    auto commit = [&](int chunk, int pbuf) {
        _Float16* dst = &Bt[pbuf][n_loc * BT_STRIDE + cb * 4];
#pragma unroll
        for (int it = 0; it < 4; ++it) {
            float xc = rx[it], a1 = r1[it], a2 = r2[it], a3 = r3[it];
            if (PREOP) {
                const int c = (chunk << 3) + cb + it * 2;    // scalar (never OOB)
                const float s = sc[c], t = sh[c];            // scalar loads
                xc = fmaxf(xc, 0.f) * s + t;
                a1 = fmaxf(a1, 0.f) * s + t;
                a2 = fmaxf(a2, 0.f) * s + t;
                a3 = fmaxf(a3, 0.f) * s + t;
            }
            v4h g;
            g[0] = (_Float16)xc;
            g[1] = (_Float16)(a1 + a2 + a3);
            g[2] = (_Float16)(fabsf(a1 - a2) + fabsf(a2 - a3) + fabsf(a3 - a1));
            g[3] = (_Float16)fmaxf(fmaxf(a1, a2), a3);
            *(v4h*)(dst + it * 8) = g;   // c_loc = cb + 2*it -> cb*4 + it*8 halves
        }
    };

    v8f acc00, acc01, acc10, acc11;
#pragma unroll
    for (int e = 0; e < 8; ++e) {
        acc00[e] = 0.f; acc01[e] = 0.f; acc10[e] = 0.f; acc11[e] = 0.f;
    }

    // A-fragment addressing: scalar tile base + loop-invariant per-lane offset
    const size_t voffA = (size_t)l16 * Kpad + half16 * 16;   // per-lane halves
    const _Float16* wtile = w16 + (size_t)m_base * Kpad;     // scalar
    const size_t tile1 = (size_t)16 * Kpad;                  // scalar row skip

    // one iteration's fragment loads + 4 WMMAs for chunk ch (buffer p)
    auto consume = [&](int ch, int p) {
        const int kk = ch << 5;
        const _Float16* wp = wtile + kk;                     // scalar advance
        v8h a0lo = *(const v8h*)(wp + voffA);
        v8h a0hi = *(const v8h*)(wp + voffA + 8);
        v8h a1lo = *(const v8h*)(wp + voffA + tile1);
        v8h a1hi = *(const v8h*)(wp + voffA + tile1 + 8);
        const _Float16* bp0 = &Bt[p][(n_sub + l16) * BT_STRIDE + half16 * 16];
        const _Float16* bp1 = bp0 + 16 * BT_STRIDE;
        v8h b00 = *(const v8h*)bp0;
        v8h b01 = *(const v8h*)(bp0 + 8);
        v8h b10 = *(const v8h*)bp1;
        v8h b11 = *(const v8h*)(bp1 + 8);

        // pin VMEM order: fragments above this point, gathers (caller) below
        __builtin_amdgcn_sched_barrier(0x7);

        v16h A0, A1, B0, B1;
#pragma unroll
        for (int e = 0; e < 8; ++e) {
            A0[e] = a0lo[e]; A0[e + 8] = a0hi[e];
            A1[e] = a1lo[e]; A1[e + 8] = a1hi[e];
            B0[e] = b00[e];  B0[e + 8] = b01[e];
            B1[e] = b10[e];  B1[e + 8] = b11[e];
        }
        acc00 = __builtin_amdgcn_wmma_f32_16x16x32_f16(false, A0, false, B0,
                                                       (short)0, acc00, false, false);
        acc01 = __builtin_amdgcn_wmma_f32_16x16x32_f16(false, A0, false, B1,
                                                       (short)0, acc01, false, false);
        acc10 = __builtin_amdgcn_wmma_f32_16x16x32_f16(false, A1, false, B0,
                                                       (short)0, acc10, false, false);
        acc11 = __builtin_amdgcn_wmma_f32_16x16x32_f16(false, A1, false, B1,
                                                       (short)0, acc11, false, false);
    };

    const int nch = Kpad >> 5;   // always >= 2 here
    gather(0);
    commit(0, 0);
    __syncthreads();

    // steady state: gather for ch+1 is UNCONDITIONAL -> single-path waits
#pragma unroll 1
    for (int ch = 0; ch < nch - 1; ++ch) {
        const int p = ch & 1;
        {
            // A/B fragment loads are inside consume(); issue gathers between
            // them and the WMMAs via the sched_barrier in consume():
            const int kk = ch << 5;
            const _Float16* wp = wtile + kk;
            v8h a0lo = *(const v8h*)(wp + voffA);
            v8h a0hi = *(const v8h*)(wp + voffA + 8);
            v8h a1lo = *(const v8h*)(wp + voffA + tile1);
            v8h a1hi = *(const v8h*)(wp + voffA + tile1 + 8);
            const _Float16* bp0 = &Bt[p][(n_sub + l16) * BT_STRIDE + half16 * 16];
            const _Float16* bp1 = bp0 + 16 * BT_STRIDE;
            v8h b00 = *(const v8h*)bp0;
            v8h b01 = *(const v8h*)(bp0 + 8);
            v8h b10 = *(const v8h*)bp1;
            v8h b11 = *(const v8h*)(bp1 + 8);

            __builtin_amdgcn_sched_barrier(0x7);
            gather(ch + 1);                      // unconditional
            __builtin_amdgcn_sched_barrier(0x7);

            v16h A0, A1, B0, B1;
#pragma unroll
            for (int e = 0; e < 8; ++e) {
                A0[e] = a0lo[e]; A0[e + 8] = a0hi[e];
                A1[e] = a1lo[e]; A1[e + 8] = a1hi[e];
                B0[e] = b00[e];  B0[e + 8] = b01[e];
                B1[e] = b10[e];  B1[e + 8] = b11[e];
            }
            acc00 = __builtin_amdgcn_wmma_f32_16x16x32_f16(false, A0, false, B0,
                                                           (short)0, acc00, false, false);
            acc01 = __builtin_amdgcn_wmma_f32_16x16x32_f16(false, A0, false, B1,
                                                           (short)0, acc01, false, false);
            acc10 = __builtin_amdgcn_wmma_f32_16x16x32_f16(false, A1, false, B0,
                                                           (short)0, acc10, false, false);
            acc11 = __builtin_amdgcn_wmma_f32_16x16x32_f16(false, A1, false, B1,
                                                           (short)0, acc11, false, false);
        }
        commit(ch + 1, p ^ 1);
        __syncthreads();
    }

    // peeled tail: consume the last chunk (no gather, no barrier needed after)
    consume(nch - 1, (nch - 1) & 1);

    // epilogue: C/D layout -> lane col = l16, VGPR r -> row r + 8*half
#pragma unroll
    for (int nf = 0; nf < 2; ++nf) {
        const int fcol = n0 + n_sub + nf * 16 + l16;
        if (fcol < F) {
            const size_t base = (size_t)b * Cout * F + fcol;
            const v8f aT0 = nf ? acc01 : acc00;
            const v8f aT1 = nf ? acc11 : acc10;
#pragma unroll
            for (int r = 0; r < 8; ++r) {
                int m0 = m_base + r + half16 * 8;
                size_t o0 = base + (size_t)m0 * F;
                size_t o1 = o0 + (size_t)16 * F;
                float v0 = aT0[r], v1 = aT1[r];
                if (RESID) {
                    v0 = fmaxf(v0 + res[o0], 0.f);
                    v1 = fmaxf(v1 + res[o1], 0.f);
                }
                out[o0] = v0; out[o1] = v1;
            }
        }
    }
}

// ---------------------------------------------------------------------------
// Per-channel BN stats over (batch=8, F) -> scale/shift.  relu_in: stats of relu(y)
// ---------------------------------------------------------------------------
__global__ __launch_bounds__(256) void stats_kernel(
    const float* __restrict__ y, const float* __restrict__ g,
    const float* __restrict__ bb, float* __restrict__ sc, float* __restrict__ sh,
    int C, int F, int relu_in)
{
    __shared__ float rs[256], rq[256];
    const int c = blockIdx.x;
    float s = 0.f, q = 0.f;
    for (int b = 0; b < 8; ++b) {
        const float* p = y + ((size_t)b * C + c) * F;
        for (int f = threadIdx.x; f < F; f += 256) {
            float v = p[f];
            if (relu_in) v = fmaxf(v, 0.f);
            s += v; q += v * v;
        }
    }
    rs[threadIdx.x] = s; rq[threadIdx.x] = q;
    __syncthreads();
    for (int st = 128; st > 0; st >>= 1) {
        if (threadIdx.x < st) {
            rs[threadIdx.x] += rs[threadIdx.x + st];
            rq[threadIdx.x] += rq[threadIdx.x + st];
        }
        __syncthreads();
    }
    if (threadIdx.x == 0) {
        float n = 8.f * (float)F;
        float m = rs[0] / n;
        float var = rq[0] / n - m * m;
        float r = rsqrtf(var + 1e-5f);
        float scale = g[c] * r;
        sc[c] = scale;
        sh[c] = bb[c] - m * scale;
    }
}

// ---------------------------------------------------------------------------
// Fused BN-affine + relu + 3-way pool mean:  out[b,c,fp] = mean_k relu(sc*z[pk]+sh)
// ---------------------------------------------------------------------------
__global__ void pool_kernel(const float* __restrict__ z, const int* __restrict__ pl,
                            const float* __restrict__ sc, const float* __restrict__ sh,
                            float* __restrict__ out, int C, int Fin, int Fout)
{
    int i = blockIdx.x * blockDim.x + threadIdx.x;
    int total = 8 * C * Fout;
    if (i >= total) return;
    int fp = i % Fout; int t = i / Fout; int c = t % C; int b = t / C;
    const float* zp = z + ((size_t)b * C + c) * Fin;
    float s = sc[c], h = sh[c], acc = 0.f;
#pragma unroll
    for (int k = 0; k < 3; ++k) {
        int idx = pl[fp * 3 + k];
        acc += fmaxf(s * zp[idx] + h, 0.f);
    }
    out[((size_t)b * C + c) * Fout + fp] = acc * (1.f / 3.f);
}

// ---------------------------------------------------------------------------
// mean over f:  xm[b*C+c] = mean_f x[b,c,f]
// ---------------------------------------------------------------------------
__global__ __launch_bounds__(256) void mean_kernel(const float* __restrict__ x,
                                                   float* __restrict__ xm, int F)
{
    __shared__ float r[256];
    const float* p = x + (size_t)blockIdx.x * F;
    float s = 0.f;
    for (int f = threadIdx.x; f < F; f += 256) s += p[f];
    r[threadIdx.x] = s;
    __syncthreads();
    for (int st = 128; st > 0; st >>= 1) {
        if (threadIdx.x < st) r[threadIdx.x] += r[threadIdx.x + st];
        __syncthreads();
    }
    if (threadIdx.x == 0) xm[blockIdx.x] = r[0] / (float)F;
}

// ---------------------------------------------------------------------------
// FC head: h = relu(xm @ W1^T + b1); out = sigmoid(h @ W2^T + b2). One block.
// ---------------------------------------------------------------------------
__global__ __launch_bounds__(128) void head_kernel(
    const float* __restrict__ xm, const float* __restrict__ w1,
    const float* __restrict__ b1, const float* __restrict__ w2,
    const float* __restrict__ b2, float* __restrict__ out)
{
    __shared__ float hs[128];
    const int j = threadIdx.x;
    for (int b = 0; b < 8; ++b) {
        float a = b1[j];
        for (int c = 0; c < 256; ++c) a += xm[b * 256 + c] * w1[j * 256 + c];
        hs[j] = fmaxf(a, 0.f) * w2[j];
        __syncthreads();
        for (int st = 64; st > 0; st >>= 1) {
            if (j < st) hs[j] += hs[j + st];
            __syncthreads();
        }
        if (j == 0) out[b] = 1.f / (1.f + expf(-(hs[0] + b2[0])));
        __syncthreads();
    }
}

// ---------------------------------------------------------------------------
extern "C" void kernel_launch(void* const* d_in, const int* in_sizes, int n_in,
                              void* d_out, int out_size, void* d_ws, size_t ws_size,
                              hipStream_t stream)
{
    (void)in_sizes; (void)n_in; (void)out_size; (void)ws_size;
    static const int RESa[4] = {12000, 9000, 6000, 4000};
    static const int CHa[4]  = {9, 64, 128, 256};

    // setup_inputs() dict order
    const float* x_in  = (const float*)d_in[0];
    const int* nbr[3]  = {(const int*)d_in[1], (const int*)d_in[3], (const int*)d_in[5]};
    const int* pool[3] = {(const int*)d_in[2], (const int*)d_in[4], (const int*)d_in[6]};
    const float *wc[3], *wsw[3], *bng[3], *bnb[3], *ng[3], *nb[3];
    for (int i = 0; i < 3; ++i) {
        int base = 7 + i * 6;
        wc[i]  = (const float*)d_in[base + 0];
        wsw[i] = (const float*)d_in[base + 1];
        bng[i] = (const float*)d_in[base + 2];
        bnb[i] = (const float*)d_in[base + 3];
        ng[i]  = (const float*)d_in[base + 4];
        nb[i]  = (const float*)d_in[base + 5];
    }
    const float* fc1_w = (const float*)d_in[25];
    const float* fc1_b = (const float*)d_in[26];
    const float* fc2_w = (const float*)d_in[27];
    const float* fc2_b = (const float*)d_in[28];

    // workspace carve-up
    const size_t BUF = 49152000;  // 8*256*6000*4 bytes (max activation tensor)
    char* wsb = (char*)d_ws;
    float* P = (float*)(wsb);
    float* Q = (float*)(wsb + BUF);
    float* R = (float*)(wsb + 2 * BUF);
    _Float16* w16 = (_Float16*)(wsb + 3 * BUF);
    float* sc = (float*)(wsb + 3 * BUF + 2 * (size_t)1200128);
    float* sh = sc + 256;
    float* xm = sh + 256;

    // f16 weight region offsets (halves)
    const size_t woff_c[3] = {0, 53248, 282624};
    const size_t woff_s[3] = {4096, 86016, 413696};
    const int KpC[3] = {64, 256, 512};      // padded K for wc (cin*4 -> mult of 32)

    // convert weights (deterministic each launch)
    for (int i = 0; i < 3; ++i) {
        int Cout = CHa[i + 1];
        int KvC = CHa[i] * 4;
        int tot = Cout * KpC[i];
        cvt_kernel<<<(tot + 255) / 256, 256, 0, stream>>>(wc[i], w16 + woff_c[i],
                                                          KvC, KpC[i], Cout);
        int KpS = Cout * 4;  // already mult of 32
        for (int s = 0; s < 3; ++s) {
            int tots = Cout * KpS;
            cvt_kernel<<<(tots + 255) / 256, 256, 0, stream>>>(
                wsw[i] + (size_t)s * Cout * KpS, w16 + woff_s[i] + (size_t)s * Cout * KpS,
                KpS, KpS, Cout);
        }
    }

    const float* cur_in = x_in;
    for (int i = 0; i < 3; ++i) {
        const int Cin = CHa[i], Cout = CHa[i + 1], F = RESa[i];
        dim3 g((F + NT - 1) / NT, Cout / 64, 8);

        // y0 = mesh_conv(x_in, wc)  -> P  (residual)
        if (i == 0)  // Cin=9: zero-padded K tail -> PARTIAL (clamped) variant
            conv_kernel<0, 1, 0><<<g, 256, 0, stream>>>(
                cur_in, nbr[i], w16 + woff_c[i], P, nullptr, nullptr, nullptr,
                Cin, Cout, F, KpC[i]);
        else
            conv_kernel<0, 0, 0><<<g, 256, 0, stream>>>(
                cur_in, nbr[i], w16 + woff_c[i], P, nullptr, nullptr, nullptr,
                Cin, Cout, F, KpC[i]);

        const int KpS = Cout * 4;
        const float* c2 = P;
        for (int s = 0; s < 3; ++s) {
            stats_kernel<<<Cout, 256, 0, stream>>>(c2, bng[i] + s * Cout,
                                                   bnb[i] + s * Cout, sc, sh,
                                                   Cout, F, 1);
            float* dst = (s == 0) ? Q : (s == 1) ? R : Q;
            const _Float16* wsk = w16 + woff_s[i] + (size_t)s * Cout * KpS;
            if (s == 2)
                conv_kernel<1, 0, 1><<<g, 256, 0, stream>>>(
                    c2, nbr[i], wsk, dst, P, sc, sh, Cout, Cout, F, KpS);
            else
                conv_kernel<1, 0, 0><<<g, 256, 0, stream>>>(
                    c2, nbr[i], wsk, dst, nullptr, sc, sh, Cout, Cout, F, KpS);
            c2 = dst;
        }
        // z = relu(y3 + y0) is in Q; outer BN + relu + pool -> R
        stats_kernel<<<Cout, 256, 0, stream>>>(Q, ng[i], nb[i], sc, sh, Cout, F, 0);
        const int Fout = RESa[i + 1];
        int tot = 8 * Cout * Fout;
        pool_kernel<<<(tot + 255) / 256, 256, 0, stream>>>(Q, pool[i], sc, sh, R,
                                                           Cout, F, Fout);
        cur_in = R;
    }

    mean_kernel<<<8 * 256, 256, 0, stream>>>(cur_in, xm, RESa[3]);
    head_kernel<<<1, 128, 0, stream>>>(xm, fc1_w, fc1_b, fc2_w, fc2_b, (float*)d_out);
}
